// RecurrentStructureBackpropNetwork_13563506720938
// MI455X (gfx1250) — matrix operand order repaired and checked by reading, not compiled, not run on hardware
//
#include <hip/hip_runtime.h>
#include <math.h>

// ---------------------------------------------------------------------------
// Problem constants
// ---------------------------------------------------------------------------
#define N_IN   256
#define N_HID  512
#define N_OUT  128
#define KDIM   768          // N_IN + N_HID (masked weight rows that matter)
#define BATCH  128
#define TSTEPS 1024
#define NWG    8            // workgroups in persistent kernel
#define SAK    776          // LDS row stride in halves (768 + 8 pad -> bank-conflict free)

typedef _Float16 v8h  __attribute__((ext_vector_type(8)));
typedef _Float16 v16h __attribute__((ext_vector_type(16)));
typedef float    v8f  __attribute__((ext_vector_type(8)));

// workspace layout (bytes)
#define XH_OFF   0u                       // [T][B][256] f16  = 67,108,864
#define WH_OFF   67108864u                // [512][768] f16   =    786,432
#define WO_OFF   67895296u                // [128][768] f16   =    196,608
#define HB_OFF   68091904u                // 2 * [128][512] f16 =  262,144
#define CNT_OFF  68354048u                // int counter

__device__ __forceinline__ v16h cat16(v8h lo, v8h hi) {
    return __builtin_shufflevector(lo, hi, 0,1,2,3,4,5,6,7,8,9,10,11,12,13,14,15);
}

// tanh via TRANS unit: v_exp_f32 + v_rcp_f32 (co-executes with WMMA)
__device__ __forceinline__ float fast_tanh(float x) {
    float e = __builtin_amdgcn_exp2f(x * 2.885390081777927f);  // exp(2x)
    float r = __builtin_amdgcn_rcpf(e + 1.0f);
    return fmaf(-2.0f, r, 1.0f);
}

// ---------------------------------------------------------------------------
// Prep kernels
// ---------------------------------------------------------------------------
__global__ void init_counter(int* cnt) { if (threadIdx.x == 0) *cnt = 0; }

// WhT[n][k] = W[k][256+n], WoT[n][k] = W[k][768+n]  (k < 768), fp32 -> fp16
__global__ void cvt_weights(const float* __restrict__ w,
                            _Float16* __restrict__ whT,
                            _Float16* __restrict__ woT) {
    int id = blockIdx.x * 256 + threadIdx.x;
    if (id < N_HID * KDIM) {
        int n = id / KDIM, k = id % KDIM;
        whT[id] = (_Float16)w[k * 896 + N_IN + n];
    }
    if (id < N_OUT * KDIM) {
        int n = id / KDIM, k = id % KDIM;
        woT[id] = (_Float16)w[k * 896 + (N_IN + N_HID) + n];
    }
}

// xh[t][b][k] = x[b][t][k], fp32 -> fp16
__global__ void cvt_x(const float* __restrict__ x, _Float16* __restrict__ xh) {
    int id = blockIdx.x * 256 + threadIdx.x;     // 2^25 elements
    int k = id & 255;
    int b = (id >> 8) & 127;
    int t = id >> 15;
    xh[id] = (_Float16)x[(((long)b << 10) + t) * 256 + k];
}

// ---------------------------------------------------------------------------
// Persistent recurrent kernel
// ---------------------------------------------------------------------------
__global__ void __launch_bounds__(256, 1)
rnn_main(const _Float16* __restrict__ xh,
         const _Float16* __restrict__ whT,
         const _Float16* __restrict__ woT,
         _Float16* __restrict__ hbuf,          // 2 * [128][512]
         int* __restrict__ cnt,
         float* __restrict__ out)              // [B][T][128]
{
    extern __shared__ _Float16 smem[];
    _Float16* acts = smem;                        // 128 rows x SAK  (x_t | h)
    _Float16* lwh  = smem + 128 * SAK;            // 64 rows x SAK
    _Float16* lwo  = smem + (128 + 64) * SAK;     // 16 rows x SAK

    const int tid    = threadIdx.x;
    const int wg     = blockIdx.x;
    const int wave   = tid >> 5;
    const int lane   = tid & 31;
    const int lanelo = lane & 15;
    const int hi     = lane >> 4;                 // half-wave select

    // ---- load Wh slice: rows wg*64..+64, 4 threads/row, 192 halves each ----
    {
        int row = tid >> 2, part = (tid & 3) * 192;
        const _Float16* src = whT + (wg * 64 + row) * KDIM + part;
        _Float16* dst = lwh + row * SAK + part;
        #pragma unroll
        for (int i = 0; i < 24; ++i) ((v8h*)dst)[i] = ((const v8h*)src)[i];
    }
    // ---- load Wo slice: rows wg*16..+16, 16 threads/row, 48 halves each ----
    {
        int row = tid >> 4, part = (tid & 15) * 48;
        const _Float16* src = woT + (wg * 16 + row) * KDIM + part;
        _Float16* dst = lwo + row * SAK + part;
        #pragma unroll
        for (int i = 0; i < 6; ++i) ((v8h*)dst)[i] = ((const v8h*)src)[i];
    }
    // ---- X_0 into acts[:, 0:256], zero h region acts[:, 256:768] ----
    {
        int row = tid >> 1;
        const _Float16* src = xh + row * N_IN + (tid & 1) * 128;
        _Float16* dst = acts + row * SAK + (tid & 1) * 128;
        #pragma unroll
        for (int i = 0; i < 16; ++i) ((v8h*)dst)[i] = ((const v8h*)src)[i];
        v8h z = {};
        _Float16* hdst = acts + row * SAK + N_IN + (tid & 1) * 256;
        #pragma unroll
        for (int i = 0; i < 32; ++i) ((v8h*)hdst)[i] = z;
    }
    __syncthreads();

    // h-GEMM wave blocking: 2M x 2N per wave (4 M-blocks x 2 N-blocks = 8 waves)
    const int mBlk = wave >> 1;                   // 0..3 -> rows 32*mBlk..+32
    const int nBlk = wave & 1;                    // 0..1 -> cols 32*nBlk..+32 (within WG slice)
    const _Float16* aB0 = acts + ((2 * mBlk + 0) * 16 + lanelo) * SAK;
    const _Float16* aB1 = acts + ((2 * mBlk + 1) * 16 + lanelo) * SAK;
    const _Float16* bB0 = lwh + ((2 * nBlk + 0) * 16 + lanelo) * SAK;
    const _Float16* bB1 = lwh + ((2 * nBlk + 1) * 16 + lanelo) * SAK;
    // out-GEMM: wave = M-tile
    const _Float16* aBo = acts + (wave * 16 + lanelo) * SAK;
    const _Float16* bBo = lwo + lanelo * SAK;

    for (int t = 0; t < TSTEPS; ++t) {
        // ================= h-GEMM: 2x2 tiles, K = 768 ======================
        v8f acc[2][2] = {};
        for (int kt = 0; kt < 24; ++kt) {
            const int ka = kt * 32 + hi * 8;       // A: K ka..ka+7 & ka+16..ka+23
            const int kb = kt * 32 + hi * 8;       // B: K kb..kb+7 & kb+16..kb+23
            v16h a0 = cat16(*(const v8h*)(aB0 + ka), *(const v8h*)(aB0 + ka + 16));
            v16h a1 = cat16(*(const v8h*)(aB1 + ka), *(const v8h*)(aB1 + ka + 16));
            v16h b0 = cat16(*(const v8h*)(bB0 + kb), *(const v8h*)(bB0 + kb + 16));
            v16h b1 = cat16(*(const v8h*)(bB1 + kb), *(const v8h*)(bB1 + kb + 16));
            acc[0][0] = __builtin_amdgcn_wmma_f32_16x16x32_f16(
                false, a0, false, b0, (short)0, acc[0][0], false, false);
            acc[0][1] = __builtin_amdgcn_wmma_f32_16x16x32_f16(
                false, a0, false, b1, (short)0, acc[0][1], false, false);
            acc[1][0] = __builtin_amdgcn_wmma_f32_16x16x32_f16(
                false, a1, false, b0, (short)0, acc[1][0], false, false);
            acc[1][1] = __builtin_amdgcn_wmma_f32_16x16x32_f16(
                false, a1, false, b1, (short)0, acc[1][1], false, false);
        }

        // tanh (TRANS ops) + store new_h slice to ping-pong global buffer
        _Float16* hnew = hbuf + ((t + 1) & 1) * (BATCH * N_HID);
        #pragma unroll
        for (int mi = 0; mi < 2; ++mi) {
            #pragma unroll
            for (int ni = 0; ni < 2; ++ni) {
                #pragma unroll
                for (int j = 0; j < 8; ++j) {
                    int m = (2 * mBlk + mi) * 16 + j + hi * 8;
                    int n = wg * 64 + (2 * nBlk + ni) * 16 + lanelo;
                    hnew[m * N_HID + n] = (_Float16)fast_tanh(acc[mi][ni][j]);
                }
            }
        }

        // ================= device-wide barrier (1 per step) ================
        __threadfence();
        __syncthreads();
        if (tid == 0) {
            __hip_atomic_fetch_add(cnt, 1, __ATOMIC_ACQ_REL, __HIP_MEMORY_SCOPE_AGENT);
            const int target = NWG * (t + 1);
            while (__hip_atomic_load(cnt, __ATOMIC_ACQUIRE, __HIP_MEMORY_SCOPE_AGENT) < target)
                __builtin_amdgcn_s_sleep(1);
        }
        __syncthreads();

        // ---- reload full H_{t+1} into acts[:, 256:768] ----
        {
            int row = tid >> 1, coff = (tid & 1) * 256;
            const _Float16* src = hnew + row * N_HID + coff;
            _Float16* dst = acts + row * SAK + N_IN + coff;
            #pragma unroll
            for (int i = 0; i < 32; ++i) ((v8h*)dst)[i] = ((const v8h*)src)[i];
        }
        __syncthreads();

        // ================= out-GEMM: 1 M-tile per wave =====================
        v8f o = {};
        for (int kt = 0; kt < 24; ++kt) {
            const int ka = kt * 32 + hi * 8;
            const int kb = kt * 32 + hi * 8;
            v16h a = cat16(*(const v8h*)(aBo + ka), *(const v8h*)(aBo + ka + 16));
            v16h b = cat16(*(const v8h*)(bBo + kb), *(const v8h*)(bBo + kb + 16));
            o = __builtin_amdgcn_wmma_f32_16x16x32_f16(
                false, a, false, b, (short)0, o, false, false);
        }
        #pragma unroll
        for (int j = 0; j < 8; ++j) {
            int m = wave * 16 + j + hi * 8;                    // batch row
            out[((long)m * TSTEPS + t) * N_OUT + wg * 16 + lanelo] = o[j];
        }

        // ---- load x_{t+1} into acts[:, 0:256] (prefetch t+2 into L2) ----
        __syncthreads();
        if (t + 1 < TSTEPS) {
            int row = tid >> 1, xoff = (tid & 1) * 128;
            const _Float16* src = xh + (t + 1) * (BATCH * N_IN) + row * N_IN + xoff;
            _Float16* dst = acts + row * SAK + xoff;
            #pragma unroll
            for (int i = 0; i < 16; ++i) ((v8h*)dst)[i] = ((const v8h*)src)[i];
            if (t + 2 < TSTEPS)
                __builtin_prefetch(xh + (t + 2) * (BATCH * N_IN) + tid * 128, 0, 1);
        }
        __syncthreads();
    }
}

// ---------------------------------------------------------------------------
// Launcher
// ---------------------------------------------------------------------------
extern "C" void kernel_launch(void* const* d_in, const int* in_sizes, int n_in,
                              void* d_out, int out_size, void* d_ws, size_t ws_size,
                              hipStream_t stream) {
    const float* x = (const float*)d_in[0];     // [128][1024][256] f32
    const float* w = (const float*)d_in[1];     // [896][896] f32
    float* out = (float*)d_out;                 // [128][1024][128] f32

    char* ws = (char*)d_ws;
    _Float16* xh  = (_Float16*)(ws + XH_OFF);
    _Float16* whT = (_Float16*)(ws + WH_OFF);
    _Float16* woT = (_Float16*)(ws + WO_OFF);
    _Float16* hb  = (_Float16*)(ws + HB_OFF);
    int*      cnt = (int*)(ws + CNT_OFF);

    const size_t ldsBytes = (size_t)(128 + 64 + 16) * SAK * sizeof(_Float16); // 322,816
    (void)hipFuncSetAttribute((const void*)rnn_main,
                              hipFuncAttributeMaxDynamicSharedMemorySize,
                              (int)ldsBytes);

    init_counter<<<1, 64, 0, stream>>>(cnt);
    cvt_weights<<<(N_HID * KDIM + 255) / 256, 256, 0, stream>>>(w, whT, woT);
    cvt_x<<<(TSTEPS * BATCH * N_IN) / 256, 256, 0, stream>>>(x, xh);
    rnn_main<<<NWG, 256, ldsBytes, stream>>>(xh, whT, woT, hb, cnt, out);
}
